// SpatialTemporalConv_59468117180806
// MI455X (gfx1250) — matrix operand order repaired and checked by reading, not compiled, not run on hardware
//
#include <hip/hip_runtime.h>
#include <cstdint>
#include <cstddef>

// ---------------- CDNA5 wave32 WMMA types ----------------
typedef __attribute__((ext_vector_type(16))) _Float16 v16h;
typedef __attribute__((ext_vector_type(8)))  float    v8f;

union AF { v16h v; uint4 q[2]; };                 // 32B f16 fragment
union CF { v8f v; float4 f[2]; float s[8]; };     // 32B f32 accumulator

#define B_    32
#define C_    64
#define T_    300
#define J_    25
#define NPB   7500        // T_*J_ columns per batch
#define NTOT  240000      // B_*NPB (BN reduction count)
#define ELEMS 15360000    // NTOT*64
#define K2    576         // 64 channels * 9 taps
#define NT_   469         // ceil(NPB/16) N-tiles
#define NT2_  235         // ceil(NPB/32) N-tile pairs
#define STATS_BLOCKS 240
#define HPAD  100         // zero columns on each side of a batch slab of h
#define HCOLS 7700        // NPB + 2*HPAD
#define HELEMS (B_ * HCOLS * 64)   // 15,769,600 padded h elements

// ---------------- weight conversion fp32 -> f16, conv layout ----------------
// W1[o*64+c]           = gcn_w[o][c]
// W2[o*576 + tap*64+i] = tcn_w[o][i][tap]   (K index = tap*64 + i)
__global__ void k_wconv(const float* __restrict__ gw, const float* __restrict__ tw,
                        _Float16* __restrict__ W1, _Float16* __restrict__ W2) {
  int idx = blockIdx.x * 256 + threadIdx.x;
  if (idx < 64 * 64) W1[idx] = (_Float16)gw[idx];
  int j = idx - 64 * 64;
  if (j >= 0 && j < 64 * K2) {
    int o = j / K2, q = j - o * K2;
    int tap = q >> 6, i = q & 63;
    W2[j] = (_Float16)tw[(o * 64 + i) * 9 + tap];
  }
}

// ---------------- joint mixing: xa[b, n, c] = sum_j x[b,c,t,j] * A[j,k] ----------------
__global__ void k_joint(const float* __restrict__ x, const float* __restrict__ A,
                        _Float16* __restrict__ xa) {
  __shared__ float Ax[625];
  __shared__ float xt[64 * 101];
  int b    = blockIdx.x / 75;
  int tile = blockIdx.x % 75;
  int base = tile * 100;                      // = t0*25, 100 positions per tile
  for (int i = threadIdx.x; i < 625; i += 256) Ax[i] = A[i];
  for (int i = threadIdx.x; i < 6400; i += 256) {
    int c = i / 100, r = i - c * 100;
    xt[c * 101 + r] = x[((size_t)(b * 64 + c)) * NPB + base + r];
  }
  __syncthreads();
  for (int i = threadIdx.x; i < 6400; i += 256) {
    int c = i & 63, p = i >> 6;               // p = local (t*25 + k), 0..99
    int tl = p / 25, k = p - tl * 25;
    const float* xr = &xt[c * 101 + tl * 25];
    float acc = 0.f;
#pragma unroll
    for (int j = 0; j < 25; ++j) acc += xr[j] * Ax[j * 25 + k];
    xa[((size_t)b * NPB + base + p) * 64 + c] = (_Float16)acc;
  }
}

// ---------------- GEMM1: z[b,n,o] = sum_c W1[o,c]*xa[b,n,c] + gcn_b[o] ----------------
// Each wave owns one M-tile (m = gw&3): A fragments hoisted out of the N loop.
// Edge N-tiles: loads run unconditionally (finite in-workspace garbage, never stored);
// only the store is guarded -> no selects, no scratch, pure global_load_b128.
__global__ void k_gemm1(const _Float16* __restrict__ W1, const _Float16* __restrict__ xa,
                        const float* __restrict__ bias, float* __restrict__ z) {
  const int lane = threadIdx.x & 31;
  const int ls = lane & 15, hs = lane >> 4;
  const int gw = blockIdx.x * 8 + (threadIdx.x >> 5);
  const int nw = gridDim.x * 8;               // launch grid: waves % 4 == 0
  const int m  = gw & 3;
  const _Float16* wrow = W1 + (m * 16 + ls) * 64;
  AF a0, a1;
  a0.q[0] = *(const uint4*)(wrow + 16 * hs);
  a0.q[1] = *(const uint4*)(wrow + 16 * hs + 8);
  a1.q[0] = *(const uint4*)(wrow + 32 + 16 * hs);
  a1.q[1] = *(const uint4*)(wrow + 32 + 16 * hs + 8);
  float bv[8];
#pragma unroll
  for (int i = 0; i < 8; ++i) bv[i] = bias[m * 16 + 8 * hs + i];
  for (int p = gw >> 2; p < B_ * NT_; p += (nw >> 2)) {
    int b = p / NT_;
    int n = p - b * NT_;
    int N = n * 16 + ls;
    bool ok = (N < NPB);
    const _Float16* col = xa + ((size_t)b * NPB + N) * 64;
    AF bf;
    CF acc; acc.v = (v8f){0.f, 0.f, 0.f, 0.f, 0.f, 0.f, 0.f, 0.f};
    bf.q[0] = *(const uint4*)(col + 16 * hs);
    bf.q[1] = *(const uint4*)(col + 16 * hs + 8);
    acc.v = __builtin_amdgcn_wmma_f32_16x16x32_f16(false, a0.v, false, bf.v,
                                                   (short)0, acc.v, false, false);
    bf.q[0] = *(const uint4*)(col + 32 + 16 * hs);
    bf.q[1] = *(const uint4*)(col + 32 + 16 * hs + 8);
    acc.v = __builtin_amdgcn_wmma_f32_16x16x32_f16(false, a1.v, false, bf.v,
                                                   (short)0, acc.v, false, false);
#pragma unroll
    for (int i = 0; i < 8; ++i) acc.s[i] += bv[i];
    if (ok) {
      float* dst = z + ((size_t)b * NPB + N) * 64 + m * 16 + 8 * hs;
      *(float4*)dst       = acc.f[0];
      *(float4*)(dst + 4) = acc.f[1];
    }
  }
}

// ---------------- GEMM2: implicit temporal conv, K = (tap,cin) = 576 ----------------
// h is stored padded: per-batch slab of HCOLS columns, data at +HPAD, 100 zero
// columns each side. Every tap-shifted source column is in-slab, so out-of-range
// taps read real zeros (= conv zero padding). Inner loop: loads + WMMA only.
// Two N-tiles per A fragment (36 WMMA per task).
__global__ void k_gemm2(const _Float16* __restrict__ W2, const _Float16* __restrict__ h,
                        const float* __restrict__ bias, float* __restrict__ u) {
  const int lane = threadIdx.x & 31;
  const int ls = lane & 15, hs = lane >> 4;
  const int gw = blockIdx.x * 8 + (threadIdx.x >> 5);
  const int nw = gridDim.x * 8;               // launch grid: waves % 4 == 0
  const int m  = gw & 3;
  const _Float16* wrow = W2 + (m * 16 + ls) * K2;
  float bv[8];
#pragma unroll
  for (int i = 0; i < 8; ++i) bv[i] = bias[m * 16 + 8 * hs + i];
  for (int p = gw >> 2; p < B_ * NT2_; p += (nw >> 2)) {
    int b  = p / NT2_;
    int n2 = p - b * NT2_;
    int N0 = n2 * 32 + ls;
    int N1 = N0 + 16;
    bool ok0 = (N0 < NPB), ok1 = (N1 < NPB);
    const _Float16* hb = h + ((size_t)b * HCOLS + HPAD) * 64;  // batch slab, data origin
    CF acc0, acc1;
    acc0.v = (v8f){0.f, 0.f, 0.f, 0.f, 0.f, 0.f, 0.f, 0.f};
    acc1.v = (v8f){0.f, 0.f, 0.f, 0.f, 0.f, 0.f, 0.f, 0.f};
#pragma unroll
    for (int kk = 0; kk < 18; ++kk) {
      int kb = kk * 32;
      AF af, bf0, bf1;
      af.q[0] = *(const uint4*)(wrow + kb + 16 * hs);
      af.q[1] = *(const uint4*)(wrow + kb + 16 * hs + 8);
      int kb2 = kb + 16 * hs;                 // lane's K base
      int tap = kb2 >> 6;                     // 0..8
      int i0  = kb2 & 63;                     // 0/16/32/48
      int d   = (tap - 4) * 25;               // temporal shift in columns
      const _Float16* c0 = hb + (N0 + d) * 64 + i0;   // always in-slab (pads are zero)
      const _Float16* c1 = hb + (N1 + d) * 64 + i0;
      bf0.q[0] = *(const uint4*)(c0);
      bf0.q[1] = *(const uint4*)(c0 + 8);
      bf1.q[0] = *(const uint4*)(c1);
      bf1.q[1] = *(const uint4*)(c1 + 8);
      acc0.v = __builtin_amdgcn_wmma_f32_16x16x32_f16(false, af.v, false, bf0.v,
                                                      (short)0, acc0.v, false, false);
      acc1.v = __builtin_amdgcn_wmma_f32_16x16x32_f16(false, af.v, false, bf1.v,
                                                      (short)0, acc1.v, false, false);
    }
#pragma unroll
    for (int i = 0; i < 8; ++i) { acc0.s[i] += bv[i]; acc1.s[i] += bv[i]; }
    if (ok0) {
      float* dst = u + ((size_t)b * NPB + N0) * 64 + m * 16 + 8 * hs;
      *(float4*)dst       = acc0.f[0];
      *(float4*)(dst + 4) = acc0.f[1];
    }
    if (ok1) {
      float* dst = u + ((size_t)b * NPB + N1) * 64 + m * 16 + 8 * hs;
      *(float4*)dst       = acc1.f[0];
      *(float4*)(dst + 4) = acc1.f[1];
    }
  }
}

// ---------------- BN stats: per-block partial sum / sumsq per channel (deterministic) ----------------
__global__ void k_stats(const float* __restrict__ v, float* __restrict__ partial) {
  int o = threadIdx.x & 63, g = threadIdx.x >> 6;
  float s = 0.f, ss = 0.f;
  for (int r = blockIdx.x * 4 + g; r < NTOT; r += STATS_BLOCKS * 4) {
    float xv = v[(size_t)r * 64 + o];
    s += xv; ss += xv * xv;
  }
  __shared__ float l0[256], l1[256];
  l0[threadIdx.x] = s; l1[threadIdx.x] = ss;
  __syncthreads();
  if (threadIdx.x < 64) {
    s  = l0[o] + l0[64 + o] + l0[128 + o] + l0[192 + o];
    ss = l1[o] + l1[64 + o] + l1[128 + o] + l1[192 + o];
    partial[blockIdx.x * 128 + o]      = s;
    partial[blockIdx.x * 128 + 64 + o] = ss;
  }
}

__global__ void k_bnfin(const float* __restrict__ partial, const float* __restrict__ g,
                        const float* __restrict__ bt, float* __restrict__ bn) {
  int o = threadIdx.x;
  if (o >= 64) return;
  float s = 0.f, ss = 0.f;
  for (int i = 0; i < STATS_BLOCKS; ++i) {
    s  += partial[i * 128 + o];
    ss += partial[i * 128 + 64 + o];
  }
  float mean = s * (1.0f / NTOT);
  float var  = ss * (1.0f / NTOT) - mean * mean;
  float sc = g[o] * rsqrtf(var + 1e-5f);
  bn[o] = sc;
  bn[64 + o] = bt[o] - mean * sc;
}

// ---------------- h = f16(relu(bn1(z))) into the PADDED slab layout ----------------
// Writes zeros into the HPAD guard columns of every slab each call (deterministic,
// also neutralizes the harness's one-time 0xAA poison of the workspace).
__global__ void k_h(const float* __restrict__ z, const float* __restrict__ bn,
                    _Float16* __restrict__ h) {
  __shared__ float p[128];
  if (threadIdx.x < 128) p[threadIdx.x] = bn[threadIdx.x];
  __syncthreads();
  for (size_t i = (size_t)blockIdx.x * 256 + threadIdx.x; i < (size_t)HELEMS;
       i += (size_t)gridDim.x * 256) {
    int o = (int)(i & 63);
    int g = (int)(i >> 6);            // padded column index
    int b = g / HCOLS;
    int cl = g - b * HCOLS;
    int n = cl - HPAD;                // data column within batch (may be pad)
    float v = 0.f;
    if ((unsigned)n < (unsigned)NPB) {
      float zv = z[((size_t)b * NPB + n) * 64 + o];
      v = fmaxf(zv * p[o] + p[64 + o], 0.f);
    }
    h[i] = (_Float16)v;
  }
}

// ---------------- out = relu(bn2(u) + x), [b,n,o] -> [b,o,t,j] via LDS transpose ----------------
__global__ void k_final(const float* __restrict__ u, const float* __restrict__ x,
                        const float* __restrict__ bn2, float* __restrict__ out) {
  __shared__ float tile[64 * 65];
  __shared__ float prm[128];
  if (threadIdx.x < 128) prm[threadIdx.x] = bn2[threadIdx.x];
  int b  = blockIdx.x / 118;
  int nt = blockIdx.x % 118;
  int n0 = nt * 64;
  for (int it = 0; it < 16; ++it) {
    int lin = it * 256 + threadIdx.x;        // 0..4095
    int nl = lin >> 6, o = lin & 63;         // coalesced read of u (o contiguous)
    int n = n0 + nl;
    float v = 0.f;
    if (n < NPB) v = u[((size_t)b * NPB + n) * 64 + o];
    tile[o * 65 + nl] = v;
  }
  __syncthreads();
  for (int it = 0; it < 16; ++it) {
    int lin = it * 256 + threadIdx.x;
    int o = lin >> 6, nl = lin & 63;         // coalesced write of out (n contiguous)
    int n = n0 + nl;
    if (n < NPB) {
      float v = tile[o * 65 + nl];
      v = v * prm[o] + prm[64 + o];
      size_t oi = ((size_t)b * 64 + o) * NPB + n;
      out[oi] = fmaxf(v + x[oi], 0.f);
    }
  }
}

// ---------------- host-side launch ----------------
extern "C" void kernel_launch(void* const* d_in, const int* in_sizes, int n_in,
                              void* d_out, int out_size, void* d_ws, size_t ws_size,
                              hipStream_t stream) {
  (void)in_sizes; (void)n_in; (void)out_size; (void)ws_size;
  const float* x     = (const float*)d_in[0];
  const float* A     = (const float*)d_in[1];
  const float* gcn_w = (const float*)d_in[2];
  const float* gcn_b = (const float*)d_in[3];
  const float* bn1_g = (const float*)d_in[4];
  const float* bn1_b = (const float*)d_in[5];
  const float* tcn_w = (const float*)d_in[6];
  const float* tcn_b = (const float*)d_in[7];
  const float* bn2_g = (const float*)d_in[8];
  const float* bn2_b = (const float*)d_in[9];
  float* out = (float*)d_out;

  // workspace layout (all regions fully rewritten each call; ~93.2 MB):
  char* w = (char*)d_ws;
  float*    bn1p = (float*)(w + 0);                    // 128 f
  float*    bn2p = (float*)(w + 512);                  // 128 f
  float*    part = (float*)(w + 1024);                 // 240*128 f = 122880 B
  _Float16* W1   = (_Float16*)(w + 126976);            // 8192 B
  _Float16* W2   = (_Float16*)(w + 126976 + 8192);     // 73728 B
  _Float16* bufA = (_Float16*)(w + 212992);            // xa, later padded h (31,539,200 B)
  float*    bufB = (float*)(w + 212992 + 31539200);    // z,  later u        (61,440,000 B)

  k_wconv<<<160, 256, 0, stream>>>(gcn_w, tcn_w, W1, W2);
  k_joint<<<B_ * 75, 256, 0, stream>>>(x, A, bufA);                 // xa (compact layout)
  k_gemm1<<<960, 256, 0, stream>>>(W1, bufA, gcn_b, bufB);          // z
  k_stats<<<STATS_BLOCKS, 256, 0, stream>>>(bufB, part);
  k_bnfin<<<1, 64, 0, stream>>>(part, bn1_g, bn1_b, bn1p);
  k_h<<<3850, 256, 0, stream>>>(bufB, bn1p, bufA);                  // padded h (reuse xa)
  k_gemm2<<<1920, 256, 0, stream>>>(W2, bufA, tcn_b, bufB);         // u (reuse z)
  k_stats<<<STATS_BLOCKS, 256, 0, stream>>>(bufB, part);
  k_bnfin<<<1, 64, 0, stream>>>(part, bn2_g, bn2_b, bn2p);
  k_final<<<B_ * 118, 256, 0, stream>>>(bufB, x, bn2p, out);
}
